// LSTMWithWC_24283745092032
// MI455X (gfx1250) — matrix operand order repaired and checked by reading, not compiled, not run on hardware
//
#include <hip/hip_runtime.h>

#define VOCAB 50000
#define EMB   128
#define HID   256
#define NCLS  3
#define BATCH 256
#define SEQ   512

typedef _Float16 h16 __attribute__((ext_vector_type(16)));
typedef _Float16 h8  __attribute__((ext_vector_type(8)));
typedef float    f8v __attribute__((ext_vector_type(8)));

// ---- packed B-fragment geometry -------------------------------------------
// Each fragment = 32(K) x 16(N) f16 = 512 halves, laid out per lane:
//   lane l in [0,16):  n = l,      halves i=0..15 -> k = i
//   lane l in [16,32): n = l - 16, halves i=0..15 -> k = 16 + i
#define FRAG_H   512
#define GATE_NT  64          // 1024 / 16  (gates:  [W;U] K=384, N=1024)
#define GATE_KF  12          // 384 / 32
#define PEEP_NT  32          // 512 / 16   (P: K=256, N=512)
#define PEEP_KF  8
#define PO_NT    16          // 256 / 16   (P_o: K=256, N=256)
#define PO_KF    8
#define GATES_FRAGS (GATE_NT * GATE_KF)   // 768
#define PEEP_FRAGS  (PEEP_NT * PEEP_KF)   // 256
#define PO_FRAGS    (PO_NT  * PO_KF)      // 128
#define TOTAL_FRAGS (GATES_FRAGS + PEEP_FRAGS + PO_FRAGS)  // 1152

// padded LDS strides (halves): break 64-bank conflict pattern, keep 16B align
#define XS (EMB + 8)
#define CS (HID + 8)

// ---------------------------------------------------------------------------
// Prep kernel: fp32 weights -> f16 WMMA B-fragments in workspace.
// ---------------------------------------------------------------------------
__global__ __launch_bounds__(256)
void pack_weights(const float* __restrict__ W,  const float* __restrict__ U,
                  const float* __restrict__ P,  const float* __restrict__ Po,
                  _Float16* __restrict__ out)
{
    const int fid = blockIdx.x;
    int ntile, kf, kind;
    _Float16* dst;
    if (fid < GATES_FRAGS)                 { kind = 0; ntile = fid / GATE_KF; kf = fid % GATE_KF;
                                             dst = out + fid * FRAG_H; }
    else if (fid < GATES_FRAGS+PEEP_FRAGS) { int g = fid - GATES_FRAGS; kind = 1;
                                             ntile = g / PEEP_KF; kf = g % PEEP_KF;
                                             dst = out + (GATES_FRAGS + g) * FRAG_H; }
    else                                   { int g = fid - GATES_FRAGS - PEEP_FRAGS; kind = 2;
                                             ntile = g / PO_KF; kf = g % PO_KF;
                                             dst = out + (GATES_FRAGS + PEEP_FRAGS + g) * FRAG_H; }
    #pragma unroll
    for (int q = 0; q < 2; ++q) {
        const int p    = threadIdx.x + q * 256;        // 0..511
        const int lane = p >> 4;
        const int i    = p & 15;
        const int k    = kf * 32 + ((lane & 16) ? 16 : 0) + i;
        const int n    = ntile * 16 + (lane & 15);
        float v;
        if (kind == 0) v = (k < EMB) ? W[k * (4*HID) + n] : U[(k - EMB) * (4*HID) + n];
        else if (kind == 1) v = P[k * (2*HID) + n];
        else v = Po[k * HID + n];
        dst[p] = (_Float16)v;
    }
}

// ---------------------------------------------------------------------------
// Fragment loaders
// ---------------------------------------------------------------------------
__device__ __forceinline__ h16 load_a(const _Float16* __restrict__ base, int stride,
                                      int row, int kbase, int lane)
{
    const int c0 = (lane & 16) ? 16 : 0;
    const h8 lo = *(const h8*)(base + row * stride + kbase + c0);
    const h8 hi = *(const h8*)(base + row * stride + kbase + c0 + 8);
    h16 a;
    #pragma unroll
    for (int i = 0; i < 8; ++i) { a[i] = lo[i]; a[i + 8] = hi[i]; }
    return a;
}

__device__ __forceinline__ h16 load_b(const _Float16* __restrict__ frag, int lane)
{
    const h8 lo = *(const h8*)(frag + lane * 16);
    const h8 hi = *(const h8*)(frag + lane * 16 + 8);
    h16 b;
    #pragma unroll
    for (int i = 0; i < 8; ++i) { b[i] = lo[i]; b[i + 8] = hi[i]; }
    return b;
}

#define WMMA(a, b, c) \
    __builtin_amdgcn_wmma_f32_16x16x32_f16(false, (a), false, (b), (short)0, (c), false, false)

// gfx1250 has a native v_tanh_f32 transcendental
__device__ __forceinline__ float fast_tanh(float x) {
#if __has_builtin(__builtin_amdgcn_tanhf)
    return __builtin_amdgcn_tanhf(x);
#elif __has_builtin(__builtin_amdgcn_tanh_f32)
    return __builtin_amdgcn_tanh_f32(x);
#else
    const float e = __expf(2.f * x);
    return 1.f - 2.f / (e + 1.f);
#endif
}
__device__ __forceinline__ float fast_sigmoid(float x) {
    return __builtin_fmaf(fast_tanh(0.5f * x), 0.5f, 0.5f);   // exact identity
}

// ---------------------------------------------------------------------------
// Main persistent recurrence kernel: 8 blocks x 1024 threads (32 waves).
// wave = m*16 + s : m in {0,1} = 16-row batch sub-tile, s = 16-col hidden slice.
// Each wave owns the fp32 c-tile (m, s) in registers for the whole sequence.
// ---------------------------------------------------------------------------
__global__ __launch_bounds__(1024)
void lstm_peephole_kernel(const int*   __restrict__ tokens,
                          const int*   __restrict__ lens,
                          const float* __restrict__ emb,
                          const float* __restrict__ bias,
                          const float* __restrict__ Wout,
                          const float* __restrict__ bout,
                          const _Float16* __restrict__ packed,
                          float* __restrict__ out)
{
    __shared__ __align__(16) _Float16 sX[32 * XS];     // x_t f16, ~8.5 KB
    __shared__ __align__(16) _Float16 sC[32 * CS];     // c f16 (A operand), ~16.5 KB
    __shared__ __align__(16) float    sLast[32 * HID]; // captured h rows, 32 KB

    const int tid  = threadIdx.x;
    const int lane = tid & 31;
    const int wave = tid >> 5;
    const int s    = wave & 15;
    const int m    = wave >> 4;
    const int brow0 = blockIdx.x * 32;

    for (int i = tid; i < 32 * CS; i += 1024) sC[i] = (_Float16)0.f;

    const int   coln = s * 16 + (lane & 15);           // hidden column of this lane
    const float bi = bias[coln];
    const float bf = bias[HID + coln];
    const float bg = bias[2 * HID + coln];
    const float bo = bias[3 * HID + coln];

    // last-step index for each of this lane's 8 C/D rows (fixed mapping)
    int lenv[8];
    #pragma unroll
    for (int v = 0; v < 8; ++v) {
        const int M = (lane < 16) ? v : v + 8;
        lenv[v] = lens[brow0 + m * 16 + M] - 1;
    }

    const int arow = m * 16 + (lane & 15);             // A-operand row for this lane
    f8v creg = {};                                     // fp32 cell-state tile (persistent)

    const int ldrow = tid >> 5;                        // x loader: row 0..31
    const int ldc4  = (tid & 31) * 4;                  // 4 floats per thread

    // prologue: x_0 into sX
    {
        const int tok = tokens[(brow0 + ldrow) * SEQ];
        const float* e = emb + tok * EMB + ldc4;
        _Float16* d = sX + ldrow * XS + ldc4;
        d[0] = (_Float16)e[0]; d[1] = (_Float16)e[1];
        d[2] = (_Float16)e[2]; d[3] = (_Float16)e[3];
    }
    __syncthreads();

    for (int t = 0; t < SEQ; ++t) {
        // Opaque zero offset: keeps pointer provenance (kernel arg -> global
        // address space -> global_load_b128 saddr+imm) while making the
        // fragment loads loop-variant so they are neither hoisted (-> scratch
        // spill, round 1) nor turned into FLAT through laundered u64 (round 2).
        int zoff = 0;
        asm volatile("" : "+s"(zoff));
        const _Float16* gB = packed + zoff;
        const _Float16* pB = gB + GATES_FRAGS * FRAG_H;
        const _Float16* oB = gB + (GATES_FRAGS + PEEP_FRAGS) * FRAG_H;

        // prefetch x_{t+1} into registers (consumed in phase 3)
        const int tn = (t + 1 < SEQ) ? t + 1 : SEQ - 1;
        const int ntok = tokens[(brow0 + ldrow) * SEQ + tn];
        const float* ne = emb + ntok * EMB + ldc4;
        const float nx0 = ne[0], nx1 = ne[1], nx2 = ne[2], nx3 = ne[3];

        // ---- phase 2: G = [x_t, c] @ [W;U]  and  Pp = c @ P ----
        f8v aI = {}, aF = {}, aG = {}, aO = {}, pI = {}, pF = {};
        #pragma unroll
        for (int kf = 0; kf < GATE_KF; ++kf) {
            h16 a = (kf < 4) ? load_a(sX, XS, arow, kf * 32, lane)
                             : load_a(sC, CS, arow, (kf - 4) * 32, lane);
            h16 b0 = load_b(gB + (((0 * 16 + s) * GATE_KF) + kf) * FRAG_H, lane);
            aI = WMMA(a, b0, aI);
            h16 b1 = load_b(gB + (((1 * 16 + s) * GATE_KF) + kf) * FRAG_H, lane);
            aF = WMMA(a, b1, aF);
            h16 b2 = load_b(gB + (((2 * 16 + s) * GATE_KF) + kf) * FRAG_H, lane);
            aG = WMMA(a, b2, aG);
            h16 b3 = load_b(gB + (((3 * 16 + s) * GATE_KF) + kf) * FRAG_H, lane);
            aO = WMMA(a, b3, aO);
            if (kf >= 4) {
                const int pk = kf - 4;
                h16 bp0 = load_b(pB + ((s * PEEP_KF) + pk) * FRAG_H, lane);
                pI = WMMA(a, bp0, pI);
                h16 bp1 = load_b(pB + (((16 + s) * PEEP_KF) + pk) * FRAG_H, lane);
                pF = WMMA(a, bp1, pF);
            }
        }
        __syncthreads();   // barrier A: all reads of sC (old c) and sX done

        // ---- phase 3: gates + cell update; publish c_new f16 and x_{t+1} ----
        #pragma unroll
        for (int v = 0; v < 8; ++v) {
            const float iv = fast_sigmoid(aI[v] + bi + fast_tanh(pI[v]));
            const float fv = fast_sigmoid(aF[v] + bf + fast_tanh(pF[v]));
            const float gv = fast_tanh(aG[v] + bg);
            const float cv = fv * creg[v] + iv * gv;
            creg[v] = cv;
            const int M = (lane < 16) ? v : v + 8;     // C/D layout row
            sC[(m * 16 + M) * CS + coln] = (_Float16)cv;
        }
        {
            _Float16* d = sX + ldrow * XS + ldc4;
            d[0] = (_Float16)nx0; d[1] = (_Float16)nx1;
            d[2] = (_Float16)nx2; d[3] = (_Float16)nx3;
        }
        __syncthreads();   // barrier B: c_new + x_{t+1} visible

        // ---- phase 4: Po = c_new @ P_o ----
        f8v aP = {};
        #pragma unroll
        for (int kf = 0; kf < PO_KF; ++kf) {
            h16 a = load_a(sC, CS, arow, kf * 32, lane);
            h16 b = load_b(oB + ((s * PO_KF) + kf) * FRAG_H, lane);
            aP = WMMA(a, b, aP);
        }

        // ---- phase 5: output gate, h; capture last hidden state ----
        #pragma unroll
        for (int v = 0; v < 8; ++v) {
            const float ov = fast_sigmoid(aO[v] + bo + fast_tanh(aP[v]));
            const float hv = ov * fast_tanh(creg[v]);
            if (lenv[v] == t) {
                const int M = (lane < 16) ? v : v + 8;
                sLast[(m * 16 + M) * HID + coln] = hv;
            }
        }
    }
    __syncthreads();

    // ---- classifier: logits = last @ Wout + bout (tiny) ----
    if (tid < 32 * NCLS) {
        const int r = tid / NCLS, cc = tid % NCLS;
        float sum = bout[cc];
        #pragma unroll 4
        for (int k = 0; k < HID; ++k) sum += sLast[r * HID + k] * Wout[k * NCLS + cc];
        out[(brow0 + r) * NCLS + cc] = sum;
    }
}

// ---------------------------------------------------------------------------
extern "C" void kernel_launch(void* const* d_in, const int* in_sizes, int n_in,
                              void* d_out, int out_size, void* d_ws, size_t ws_size,
                              hipStream_t stream)
{
    const int*   tokens = (const int*)  d_in[0];
    const int*   lens   = (const int*)  d_in[1];
    const float* emb    = (const float*)d_in[2];
    const float* W      = (const float*)d_in[3];
    const float* U      = (const float*)d_in[4];
    const float* P      = (const float*)d_in[5];
    const float* Po     = (const float*)d_in[6];
    const float* bias   = (const float*)d_in[7];
    const float* Wout   = (const float*)d_in[8];
    const float* bout   = (const float*)d_in[9];
    _Float16*    packed = (_Float16*)   d_ws;    // needs 1152*512*2 = 1.125 MB
    float*       out    = (float*)      d_out;

    pack_weights<<<dim3(TOTAL_FRAGS), dim3(256), 0, stream>>>(W, U, P, Po, packed);
    lstm_peephole_kernel<<<dim3(BATCH / 32), dim3(1024), 0, stream>>>(
        tokens, lens, emb, bias, Wout, bout, packed, out);
}
